// Attention_79817672229574
// MI455X (gfx1250) — compile-verified, matrix operand order checked by reading, not run on hardware
//
#include <hip/hip_runtime.h>

// ---------------------------------------------------------------------------
// Problem constants (from the reference)
// ---------------------------------------------------------------------------
#define CDIM 512
#define NHEADS 8
#define BATCH 8
#define HW 4096              // 64*64
#define DHEAD 64             // CDIM / NHEADS
#define KCH 64               // attention K-chunk (two WMMA k-steps)

typedef __attribute__((ext_vector_type(16))) __bf16 v16bf;
typedef __attribute__((ext_vector_type(8)))  __bf16 v8bf;
typedef __attribute__((ext_vector_type(8)))  float  v8f;
typedef __attribute__((ext_vector_type(4)))  unsigned int u32x4;
typedef __attribute__((ext_vector_type(8)))  int i32x8;
typedef __attribute__((ext_vector_type(4)))  int i32x4;

#if defined(__HIP_DEVICE_COMPILE__) && \
    __has_builtin(__builtin_amdgcn_tensor_load_to_lds) && \
    __has_builtin(__builtin_amdgcn_s_wait_tensorcnt)
#define USE_TDM 1
#else
#define USE_TDM 0
#endif

// ---------------------------------------------------------------------------
// WMMA helper: D(16x16 f32) += A(16x32 bf16) * B(32x16 bf16)
// ---------------------------------------------------------------------------
static __device__ inline v8f wmma_bf16(v16bf a, v16bf b, v8f c) {
    return __builtin_amdgcn_wmma_f32_16x16x32_bf16(
        /*neg_a=*/false, a, /*neg_b=*/false, b,
        /*c_mod=*/(short)0, c, /*reuse_a=*/false, /*reuse_b=*/false);
}

// Build a 16x32 operand fragment from a row-major bf16 LDS tile.
// ISA 7.12.2 16-bit A layout: lane L (<16) holds row L, K in {klo..klo+7} u
// {klo+16..klo+23}, klo = 8*(L>=16).  Two 16B chunks -> two ds_load_b128.
// B operand uses the symmetric layout when the tile is stored as Bt[n][k].
static __device__ inline v16bf frag_ld(const __bf16* base, int stride, int lane) {
    const int row = lane & 15;
    const int klo = (lane >> 4) << 3;          // 0 or 8
    const __bf16* p = base + row * stride + klo;
    v8bf lo = *(const v8bf*)p;                 // K = klo .. klo+7
    v8bf hi = *(const v8bf*)(p + 16);          // K = klo+16 .. klo+23
    v16bf r;
#pragma unroll
    for (int i = 0; i < 8; ++i) { r[i] = lo[i]; r[i + 8] = hi[i]; }
    return r;
}

#if USE_TDM
// ---------------------------------------------------------------------------
// Tensor Data Mover: 2D bf16 tile (tile_d0 x tile_d1) from global -> LDS.
// Descriptor per ISA 08_async_tensor §8: group0 = {count, lds_addr,
// global_addr, type=2}; group1 = {data_size=2B, tensor dims, tile dims,
// dim0 stride}.  2D tensor -> groups 2/3 zero (VADDR2/3 semantics).
// 6-arg builtin form: (g0, g1, g2, g3, g_extra, cpol).
// ---------------------------------------------------------------------------
static __device__ inline void tdm_load_2d(const void* gaddr, unsigned lds_off,
                                          unsigned tile_d0, unsigned tile_d1,
                                          unsigned tensor_d0, unsigned tensor_d1,
                                          unsigned stride_e) {
    unsigned long long ga = (unsigned long long)gaddr;
    u32x4 g0;
    g0[0] = 1u;                                        // count=1, user desc
    g0[1] = lds_off;                                   // lds_addr [63:32]
    g0[2] = (unsigned)(ga & 0xffffffffu);              // global_addr lo
    g0[3] = (unsigned)((ga >> 32) & 0x01ffffffu) | (2u << 30);  // hi | type=2
    i32x8 g1;
    g1[0] = (int)(1u << 16);                           // data_size = 2 bytes
    g1[1] = (int)((tensor_d0 & 0xffffu) << 16);        // tensor_dim0[15:0]
    g1[2] = (int)(((tensor_d0 >> 16) & 0xffffu) |
                  ((tensor_d1 & 0xffffu) << 16));      // dim0 hi | dim1 lo
    g1[3] = (int)(((tensor_d1 >> 16) & 0xffffu) |
                  ((tile_d0 & 0xffffu) << 16));        // dim1 hi | tile_dim0
    g1[4] = (int)(tile_d1 & 0xffffu);                  // tile_dim1 (tile_dim2=0)
    g1[5] = (int)stride_e;                             // tensor_dim0_stride lo
    g1[6] = 0;                                         // stride hi | dim1_stride
    g1[7] = 0;
    i32x4 z4; z4[0] = 0; z4[1] = 0; z4[2] = 0; z4[3] = 0;
    i32x8 z8;
#pragma unroll
    for (int i = 0; i < 8; ++i) z8[i] = 0;
    __builtin_amdgcn_tensor_load_to_lds(g0, g1, z4, z4, z8, 0);
}
#endif

// ---------------------------------------------------------------------------
// Kernel 1/4: pointwise (1x1 conv) GEMM.  Y[z][b][m][n] = sum_c W[z][m][c] X[b][c][n]
// 128x128 block tile, 8 waves x (one 16-row strip * eight 16-col tiles),
// K stepped by 32, LDS double buffered, bf16 WMMA, f32 accumulate.
// ---------------------------------------------------------------------------
template <typename TIn, typename TOut>
__global__ __launch_bounds__(256)
void gemm_pw(const float* __restrict__ W0, const float* __restrict__ W1,
             const float* __restrict__ W2, const TIn* __restrict__ X,
             TOut* __restrict__ Y, unsigned long long ystride_z) {
    const int z  = blockIdx.z;
    const float* W = (z == 0) ? W0 : (z == 1) ? W1 : W2;
    const int bx = blockIdx.x;                 // b*32 + n-block
    const int b  = bx >> 5;
    const int n0 = (bx & 31) * 128;
    const int m0 = blockIdx.y * 128;
    const TIn* Xb = X + (size_t)b * CDIM * HW;
    TOut* Yb = Y + (size_t)z * ystride_z + (size_t)b * CDIM * HW;

    __shared__ __bf16 As[2][128][32];          // W tile  [m][k]
    __shared__ __bf16 Bs[2][128][32];          // X tile transposed: Bt[n][k]

    const int tid  = threadIdx.x;
    const int lane = tid & 31;
    const int wv   = tid >> 5;

    v8f acc[8];
#pragma unroll
    for (int i = 0; i < 8; ++i) acc[i] = (v8f)0.0f;

    auto loadA = [&](int it, int buf) {
        const int kc = it * 32;
#pragma unroll
        for (int rep = 0; rep < 2; ++rep) {
            const int idx = tid + rep * 256;
            const int row = idx >> 2, seg = idx & 3;
            const float* p = W + (size_t)(m0 + row) * CDIM + kc + seg * 8;
            float4 f0 = *(const float4*)p;
            float4 f1 = *(const float4*)(p + 4);
            v8bf o;
            o[0] = (__bf16)f0.x; o[1] = (__bf16)f0.y;
            o[2] = (__bf16)f0.z; o[3] = (__bf16)f0.w;
            o[4] = (__bf16)f1.x; o[5] = (__bf16)f1.y;
            o[6] = (__bf16)f1.z; o[7] = (__bf16)f1.w;
            *(v8bf*)&As[buf][row][seg * 8] = o;
        }
    };
    auto loadB = [&](int it, int buf) {
        const int kc = it * 32;
        const int cr = tid >> 3, l8 = tid & 7; // 8 threads per k-row
        const TIn* p = Xb + (size_t)(kc + cr) * HW + n0;
        __builtin_prefetch(p + 32 * (size_t)HW, 0, 0);  // next K chunk
#pragma unroll
        for (int j = 0; j < 4; ++j) {
            const int nf = (l8 + j * 8) * 4;
#pragma unroll
            for (int u = 0; u < 4; ++u)
                Bs[buf][nf + u][cr] = (__bf16)(float)p[nf + u];
        }
    };

    loadA(0, 0); loadB(0, 0);
    __syncthreads();
    int buf = 0;
    for (int it = 0; it < CDIM / 32; ++it) {
        if (it + 1 < CDIM / 32) { loadA(it + 1, buf ^ 1); loadB(it + 1, buf ^ 1); }
        v16bf af = frag_ld(&As[buf][wv * 16][0], 32, lane);
#pragma unroll
        for (int nt = 0; nt < 8; ++nt) {
            v16bf bfr = frag_ld(&Bs[buf][nt * 16][0], 32, lane);
            acc[nt] = wmma_bf16(af, bfr, acc[nt]);
        }
        __syncthreads();
        buf ^= 1;
    }

    // C/D layout: element (vgpr i, lane L) = row m0+wv*16+i+8*(L>=16),
    // col n0+nt*16+(L&15).
#pragma unroll
    for (int nt = 0; nt < 8; ++nt) {
#pragma unroll
        for (int i = 0; i < 8; ++i) {
            const int row = m0 + wv * 16 + i + ((lane >> 4) << 3);
            const int col = n0 + nt * 16 + (lane & 15);
            Yb[(size_t)row * HW + col] = (TOut)acc[nt][i];
        }
    }
}

// ---------------------------------------------------------------------------
// Kernel 2/4: depthwise 3x3, pad 1.  One thread per output pixel. bf16 io.
// ---------------------------------------------------------------------------
__global__ __launch_bounds__(256)
void dwconv3x3(const __bf16* __restrict__ in, const float* __restrict__ dq,
               const float* __restrict__ dk, const float* __restrict__ dv,
               __bf16* __restrict__ out) {
    const int z = blockIdx.y;
    const float* Wd = (z == 0) ? dq : (z == 1) ? dk : dv;
    const size_t zoff = (size_t)z * BATCH * CDIM * HW;
    const int idx = blockIdx.x * 256 + threadIdx.x;     // b*c*hw index
    const int bc  = idx >> 12;
    const int pix = idx & 4095;
    const int y = pix >> 6, x = pix & 63;
    const int c = bc & (CDIM - 1);
    const float* wk = Wd + c * 9;
    const __bf16* s = in + zoff + (size_t)bc * HW;
    float accv = 0.0f;
#pragma unroll
    for (int ky = 0; ky < 3; ++ky) {
        const int yy = y + ky - 1;
        if (yy < 0 || yy > 63) continue;
#pragma unroll
        for (int kx = 0; kx < 3; ++kx) {
            const int xx = x + kx - 1;
            if (xx < 0 || xx > 63) continue;
            accv += wk[ky * 3 + kx] * (float)s[yy * 64 + xx];
        }
    }
    out[zoff + (size_t)idx] = (__bf16)accv;
}

// ---------------------------------------------------------------------------
// Kernel 3/4: channel attention for one (b,h).
//  Pass 1: stream Q,K [64][4096] in 64x64 chunks (TDM double buffered) ->
//          S = Q K^T via WMMA; row sum-of-squares read back from LDS.
//  Scale by temperature / (max(|q|,eps)*max(|k|,eps)), softmax in LDS.
//  Pass 2: out = P V, staging V chunks transposed in LDS.
// ---------------------------------------------------------------------------
struct AttnLDS {
    __bf16 Qc[2][64][KCH];     // 16 KB
    __bf16 Kc[2][64][KCH];     // 16 KB
    float  Smat[64][64];       // 16 KB
    float  qq[64];
    float  kk[64];
    __bf16 Pt[64][64];         // 8 KB
    __bf16 Vt[64][64];         // 8 KB
};

__global__ __launch_bounds__(256)
void xca_attn(const __bf16* __restrict__ Qg, const __bf16* __restrict__ Kg,
              const __bf16* __restrict__ Vg, const float* __restrict__ temp,
              __bf16* __restrict__ Og) {
    const int b = blockIdx.x >> 3, h = blockIdx.x & 7;
    const size_t base = ((size_t)b * CDIM + h * DHEAD) * HW;
    const __bf16* Q = Qg + base;
    const __bf16* K = Kg + base;
    const __bf16* V = Vg + base;
    __bf16* O = Og + base;

    __shared__ AttnLDS L;

    const int tid = threadIdx.x, lane = tid & 31, wv = tid >> 5;
    if (tid < 64) { L.qq[tid] = 0.0f; L.kk[tid] = 0.0f; }

    const int r = tid >> 2, sg = tid & 3;      // 4 threads per row for staging

#if USE_TDM
    // Wave 0 drives the Tensor Data Mover; everyone else just consumes LDS.
    auto issue = [&](int it, int buf) {
        if (wv == 0) {
            const unsigned qoff = (unsigned)(__builtin_offsetof(AttnLDS, Qc) +
                                             (unsigned)buf * 64u * KCH * 2u);
            const unsigned koff = (unsigned)(__builtin_offsetof(AttnLDS, Kc) +
                                             (unsigned)buf * 64u * KCH * 2u);
            tdm_load_2d(Q + (size_t)it * KCH, qoff, KCH, 64, HW, 64, HW);
            tdm_load_2d(K + (size_t)it * KCH, koff, KCH, 64, HW, 64, HW);
        }
    };
    auto drain = [&]() { __builtin_amdgcn_s_wait_tensorcnt((short)0); };
#else
    auto issue = [&](int it, int buf) {
        const size_t go = (size_t)r * HW + (size_t)it * KCH + sg * 8;
        *(v8bf*)&L.Qc[buf][r][sg * 8]      = *(const v8bf*)(Q + go);
        *(v8bf*)&L.Qc[buf][r][sg * 8 + 32] = *(const v8bf*)(Q + go + 32);
        *(v8bf*)&L.Kc[buf][r][sg * 8]      = *(const v8bf*)(K + go);
        *(v8bf*)&L.Kc[buf][r][sg * 8 + 32] = *(const v8bf*)(K + go + 32);
    };
    auto drain = [&]() {};
#endif

    // 16 S-tiles (4x4); wave owns (mt, nt0) and (mt, nt0+1)
    const int mt = wv >> 1, nt0 = (wv & 1) * 2;
    v8f acc0 = (v8f)0.0f, acc1 = (v8f)0.0f;
    float aq = 0.0f, ak = 0.0f;

    issue(0, 0);
    drain();
    __syncthreads();
    int buf = 0;
    for (int it = 0; it < HW / KCH; ++it) {
        if (it + 1 < HW / KCH) issue(it + 1, buf ^ 1);
        // fused sum-of-squares (read back from LDS: data is already on-chip)
        {
            v8bf a = *(const v8bf*)&L.Qc[buf][r][sg * 8];
            v8bf c = *(const v8bf*)&L.Qc[buf][r][sg * 8 + 32];
            v8bf d = *(const v8bf*)&L.Kc[buf][r][sg * 8];
            v8bf e = *(const v8bf*)&L.Kc[buf][r][sg * 8 + 32];
#pragma unroll
            for (int u = 0; u < 8; ++u) {
                float fa = (float)a[u], fc = (float)c[u];
                float fd = (float)d[u], fe = (float)e[u];
                aq += fa * fa + fc * fc;
                ak += fd * fd + fe * fe;
            }
        }
#pragma unroll
        for (int ks = 0; ks < KCH; ks += 32) {
            v16bf a  = frag_ld(&L.Qc[buf][mt * 16][ks], KCH, lane);
            v16bf b0 = frag_ld(&L.Kc[buf][nt0 * 16][ks], KCH, lane);
            v16bf b1 = frag_ld(&L.Kc[buf][(nt0 + 1) * 16][ks], KCH, lane);
            acc0 = wmma_bf16(a, b0, acc0);
            acc1 = wmma_bf16(a, b1, acc1);
        }
        drain();                 // next chunk's TDM done before flipping
        __syncthreads();
        buf ^= 1;
    }
    atomicAdd(&L.qq[r], aq);
    atomicAdd(&L.kk[r], ak);
    __syncthreads();
    if (tid < 64) {   // inverse norms (F.normalize eps semantics)
        L.qq[tid] = 1.0f / fmaxf(sqrtf(L.qq[tid]), 1e-12f);
        L.kk[tid] = 1.0f / fmaxf(sqrtf(L.kk[tid]), 1e-12f);
    }
    __syncthreads();

    const float tmprt = temp[h];
#pragma unroll
    for (int i = 0; i < 8; ++i) {
        const int row = mt * 16 + i + ((lane >> 4) << 3);
        const int c0  = nt0 * 16 + (lane & 15);
        L.Smat[row][c0]      = acc0[i] * tmprt * L.qq[row] * L.kk[c0];
        L.Smat[row][c0 + 16] = acc1[i] * tmprt * L.qq[row] * L.kk[c0 + 16];
    }
    __syncthreads();

    if (tid < 64) {   // row softmax, write P as bf16 (A operand of pass 2)
        float m = -3.4e38f;
        for (int j = 0; j < 64; ++j) m = fmaxf(m, L.Smat[tid][j]);
        float s = 0.0f;
        for (int j = 0; j < 64; ++j) {
            float e = __expf(L.Smat[tid][j] - m);
            L.Smat[tid][j] = e; s += e;
        }
        const float inv = 1.0f / s;
        for (int j = 0; j < 64; ++j) L.Pt[tid][j] = (__bf16)(L.Smat[tid][j] * inv);
    }
    __syncthreads();

    // Pass 2: out[d][n] = sum_e P[d][e] V[e][n], n chunked by 64.
    for (int nc = 0; nc < HW; nc += 64) {
        {   // stage V chunk transposed: Vt[n][e] = V[e][nc+n]
            const int e = tid >> 2, s0 = tid & 3;
#pragma unroll
            for (int half = 0; half < 2; ++half) {
                const int sgg = s0 + half * 4;
                v8bf v8 = *(const v8bf*)(V + (size_t)e * HW + nc + sgg * 8);
#pragma unroll
                for (int u = 0; u < 8; ++u) L.Vt[sgg * 8 + u][e] = v8[u];
            }
        }
        __syncthreads();
        v8f o0 = (v8f)0.0f, o1 = (v8f)0.0f;
#pragma unroll
        for (int ks = 0; ks < 64; ks += 32) {
            v16bf a  = frag_ld(&L.Pt[mt * 16][ks], 64, lane);
            v16bf b0 = frag_ld(&L.Vt[nt0 * 16][ks], 64, lane);
            v16bf b1 = frag_ld(&L.Vt[(nt0 + 1) * 16][ks], 64, lane);
            o0 = wmma_bf16(a, b0, o0);
            o1 = wmma_bf16(a, b1, o1);
        }
#pragma unroll
        for (int i = 0; i < 8; ++i) {
            const int row = mt * 16 + i + ((lane >> 4) << 3);
            const int c0  = nc + nt0 * 16 + (lane & 15);
            O[(size_t)row * HW + c0]      = (__bf16)o0[i];
            O[(size_t)row * HW + c0 + 16] = (__bf16)o1[i];
        }
        __syncthreads();
    }
}

// ---------------------------------------------------------------------------
// Launch
// ---------------------------------------------------------------------------
extern "C" void kernel_launch(void* const* d_in, const int* in_sizes, int n_in,
                              void* d_out, int out_size, void* d_ws, size_t ws_size,
                              hipStream_t stream) {
    (void)in_sizes; (void)n_in; (void)out_size; (void)ws_size;
    const float* x      = (const float*)d_in[0];
    const float* w_q    = (const float*)d_in[1];
    const float* w_k    = (const float*)d_in[2];
    const float* w_v    = (const float*)d_in[3];
    const float* dw_q   = (const float*)d_in[4];
    const float* dw_k   = (const float*)d_in[5];
    const float* dw_v   = (const float*)d_in[6];
    const float* w_proj = (const float*)d_in[7];
    const float* temp   = (const float*)d_in[8];
    float* out = (float*)d_out;

    const unsigned long long BCHW = (unsigned long long)BATCH * CDIM * HW;
    char* ws = (char*)d_ws;
    __bf16* pre = (__bf16*)ws;                       // 3*BCHW bf16 (96 MB)
    __bf16* dwb = (__bf16*)(ws + 3 * BCHW * 2);      // 3*BCHW bf16 (96 MB)
    __bf16* att = pre;                               // reuse dead q_pre region

    // 1) QKV pointwise GEMMs (z = 0,1,2), bf16 out
    gemm_pw<float, __bf16><<<dim3(256, 4, 3), 256, 0, stream>>>(
        w_q, w_k, w_v, x, pre, BCHW);

    // 2) depthwise 3x3 for q,k,v
    dwconv3x3<<<dim3((unsigned)(BCHW / 256), 3), 256, 0, stream>>>(
        pre, dw_q, dw_k, dw_v, dwb);

    // 3) channel attention per (b,h)
    xca_attn<<<dim3(BATCH * NHEADS), 256, 0, stream>>>(
        dwb, dwb + BCHW, dwb + 2 * BCHW, temp, att);

    // 4) output projection, f32 out
    gemm_pw<__bf16, float><<<dim3(256, 4, 1), 256, 0, stream>>>(
        w_proj, w_proj, w_proj, att, out, 0ULL);
}
